// TemplateAttention_33535104647893
// MI455X (gfx1250) — matrix operand order repaired and checked by reading, not compiled, not run on hardware
//
#include <hip/hip_runtime.h>
#include <stdint.h>

// ---------------------------------------------------------------------------
// Template pointwise attention (AF2) for MI455X / gfx1250.
//  B=1, N=384, T=4, CQ=128, CT=64, H=4, D=16.
//  - All channel contractions on v_wmma_f32_16x16x32_bf16 (f32 softmax/accum)
//  - Tiles staged to LDS via global_load_async_to_lds_b128 (ASYNCcnt),
//    pipelined against the output projection of the previous tile
//  - Online softmax over T=4; waves_per_eu(2,4) so RA has headroom (no spills)
// ---------------------------------------------------------------------------

typedef __attribute__((ext_vector_type(16))) __bf16 v16bf;
typedef __attribute__((ext_vector_type(2)))  __bf16 v2bf;
typedef __attribute__((ext_vector_type(8)))  float  v8f;
typedef __attribute__((ext_vector_type(2)))  float  v2f;

#define N_SEQ   384
#define P_TOT   (N_SEQ * N_SEQ)   // 147456 pair positions
#define CQ      128
#define CT      64
#define NH      4
#define HD      16
#define NT      4
#define WAVES   4                 // waves per block (wave32)
#define TPW     8                 // 16-position tiles per wave
#define FP16_HUGE 65504.0f

// --- LDS layout (byte offsets into dynamic smem) ---------------------------
#define SOFF_QW   0u              // bf16 qwT [NH][16][128]      16384 B
#define SOFF_KW   16384u          // bf16 kwT [NH][16][64]        8192 B
#define SOFF_VW   24576u          // bf16 vwT [NH][16][64]        8192 B
#define SOFF_OW   32768u          // bf16 owT [128][64]          16384 B
#define SOFF_QT   49152u          // f32  Qtile  [WAVES][16][128] 8192 B/wave
#define SOFF_TT   81920u          // f32  Ttiles [WAVES][NT][16][64] 16384 B/wave
#define SOFF_WV   147456u         // bf16 wv [WAVES][16][64]      2048 B/wave
#define SMEM_BYTES 155648u        // 152 KB -> 2 workgroups / WGP (320 KB LDS)

// low 32 bits of a generic pointer to LDS == byte offset within wave's LDS
__device__ __forceinline__ uint32_t lds_off32(const void* p) {
    return (uint32_t)(uintptr_t)p;
}

// CDNA5 async copy global -> LDS (16 bytes per lane), tracked by ASYNCcnt
__device__ __forceinline__ void async_copy_b128(uint32_t lds_byte, const void* gaddr) {
    asm volatile("global_load_async_to_lds_b128 %0, %1, off"
                 :: "v"(lds_byte), "v"(gaddr)
                 : "memory");
}
__device__ __forceinline__ void wait_async0() {
    asm volatile("s_wait_asynccnt 0x0" ::: "memory");
}
__device__ __forceinline__ void wait_ds0() {
    asm volatile("s_wait_dscnt 0x0" ::: "memory");
}

__device__ __forceinline__ v8f wmma_bf16(v16bf a, v16bf b, v8f c) {
    return __builtin_amdgcn_wmma_f32_16x16x32_bf16(
        /*neg_a=*/false, a, /*neg_b=*/false, b,
        /*c_mod=*/(short)0, c, /*reuse_a=*/false, /*reuse_b=*/false);
}

// stage one 16-position tile: Q tile (8 KB) + NT template tiles (4 KB each).
// Rolled loops: keep only a couple of live address registers.
__device__ __forceinline__ void stage_tile(const float* __restrict__ qe,
                                           const float* __restrict__ tpl,
                                           int p0, uint32_t qdst, uint32_t tdst,
                                           int lane) {
    {
        const char* qsrc = (const char*)(qe + (size_t)p0 * CQ) + lane * 16;
        uint32_t    d    = qdst + (uint32_t)(lane * 16);
#pragma unroll 1
        for (int i = 0; i < 16; ++i) {
            async_copy_b128(d, qsrc);
            d += 512u;
            qsrc += 512;
        }
    }
#pragma unroll 1
    for (int t = 0; t < NT; ++t) {
        const char* tsrc = (const char*)(tpl + ((size_t)t * P_TOT + p0) * CT) + lane * 16;
        uint32_t    d    = tdst + (uint32_t)(t * 4096 + lane * 16);
#pragma unroll 1
        for (int i = 0; i < 8; ++i) {
            async_copy_b128(d, tsrc);
            d += 512u;
            tsrc += 512;
        }
    }
}

__global__ void
__launch_bounds__(WAVES * 32)
__attribute__((amdgpu_waves_per_eu(2, 4)))
tpa_wmma_kernel(const float* __restrict__ qe,    // [N,N,CQ]
                const float* __restrict__ tpl,   // [NT,N,N,CT]
                const float* __restrict__ tmsk,  // [NT]
                const float* __restrict__ qw,    // [NH,CQ,HD]
                const float* __restrict__ kw,    // [NH,CT,HD]
                const float* __restrict__ vw,    // [NH,CT,HD]
                const float* __restrict__ ow,    // [NH,HD,CQ]
                const float* __restrict__ ob,    // [CQ]
                float* __restrict__ out)         // [N,N,CQ]
{
    extern __shared__ char smem[];
    __bf16* s_qw = (__bf16*)(smem + SOFF_QW);   // [h][d][c]  (B-major: row=d, K=c)
    __bf16* s_kw = (__bf16*)(smem + SOFF_KW);
    __bf16* s_vw = (__bf16*)(smem + SOFF_VW);
    __bf16* s_ow = (__bf16*)(smem + SOFF_OW);   // [c_out][h*16+d]

    const int tid  = threadIdx.x;               // 0..127
    const int wave = tid >> 5;
    const int lane = tid & 31;
    const int g    = lane >> 4;                 // half-wave group (0/1)
    const int nm   = lane & 15;                 // M (A/C rows) or N (B/C cols)

    float*  s_q  = (float*)(smem + SOFF_QT + (uint32_t)wave * 8192u);   // [16][128]
    float*  s_t  = (float*)(smem + SOFF_TT + (uint32_t)wave * 16384u);  // [NT][16][64]
    __bf16* s_wv = (__bf16*)(smem + SOFF_WV + (uint32_t)wave * 2048u);  // [16][64]

    // ---- stage weights: transpose + convert f32 -> bf16 (once per block) ----
#pragma unroll 1
    for (int idx = tid; idx < NH * CQ * HD; idx += WAVES * 32) {
        int h = idx >> 11, rem = idx & 2047, c = rem >> 4, d = rem & 15;
        s_qw[(h * 16 + d) * 128 + c] = (__bf16)qw[idx];
    }
#pragma unroll 1
    for (int idx = tid; idx < NH * CT * HD; idx += WAVES * 32) {
        int h = idx >> 10, rem = idx & 1023, c = rem >> 4, d = rem & 15;
        s_kw[(h * 16 + d) * 64 + c] = (__bf16)kw[idx];
        s_vw[(h * 16 + d) * 64 + c] = (__bf16)vw[idx];
    }
#pragma unroll 1
    for (int idx = tid; idx < NH * HD * CQ; idx += WAVES * 32) {
        int hd = idx >> 7, c = idx & 127;
        s_ow[c * 64 + hd] = (__bf16)ow[idx];
    }

    // ---- mask -> additive bias + global gate --------------------------------
    float bias_t[NT];
    float msum = 0.f;
#pragma unroll
    for (int t = 0; t < NT; ++t) {
        float m = tmsk[t];
        bias_t[t] = FP16_HUGE * (m - 1.0f);
        msum += m;
    }
    const float gate = (msum > 0.0f) ? 1.0f : 0.0f;

    __syncthreads();

    const uint32_t qdst = lds_off32(s_q);
    const uint32_t tdst = lds_off32(s_t);
    const int tile_base = blockIdx.x * (WAVES * TPW);

    // prologue: stage first tile
    stage_tile(qe, tpl, (tile_base + wave) * 16, qdst, tdst, lane);

#pragma unroll 1
    for (int it = 0; it < TPW; ++it) {
        const int p0 = (tile_base + it * WAVES + wave) * 16;

        wait_async0();      // current tile's Q + T tiles are resident in LDS

        // ---- Q A-fragments (held across heads; 4 x v16bf = 32 VGPRs) ------
        // A 16x32 bf16: lane (g,m); pair j -> K0 = 32c + 16g + 2j
        v16bf aq[4];
#pragma unroll
        for (int c = 0; c < 4; ++c) {
#pragma unroll
            for (int j = 0; j < 8; ++j) {
                const int K0 = 32 * c + 16 * g + 2 * j;
                v2f p = *(const v2f*)(s_q + nm * 128 + K0);
                aq[c][2 * j]     = (__bf16)p.x;
                aq[c][2 * j + 1] = (__bf16)p.y;
            }
        }

        // ---- attention per head (online softmax over T) --------------------
#pragma unroll 1
        for (int h = 0; h < NH; ++h) {
            // Q projection: 4 WMMAs over K=128 (B: lane=n, K = 32c + 16g + 2j)
            v8f qacc = {0.f, 0.f, 0.f, 0.f, 0.f, 0.f, 0.f, 0.f};
#pragma unroll
            for (int c = 0; c < 4; ++c) {
                v16bf b;
#pragma unroll
                for (int j = 0; j < 8; ++j) {
                    const int k = 32 * c + 16 * g + 2 * j;
                    v2bf pr = *(const v2bf*)(s_qw + (h * 16 + nm) * 128 + k);
                    b[2 * j]     = pr.x;
                    b[2 * j + 1] = pr.y;
                }
                qacc = wmma_bf16(aq[c], b, qacc);
            }
#pragma unroll
            for (int r = 0; r < 8; ++r) qacc[r] *= 0.25f;  // d^-0.5, d=16

            // online softmax state (per C-row r): running max / sum / weighted V
            float mx[8], sm[8];
            v8f   wacc = {0.f, 0.f, 0.f, 0.f, 0.f, 0.f, 0.f, 0.f};
#pragma unroll
            for (int r = 0; r < 8; ++r) { mx[r] = -3.0e38f; sm[r] = 0.f; }

#pragma unroll 1
            for (int t = 0; t < NT; ++t) {
                v8f kacc = {0.f, 0.f, 0.f, 0.f, 0.f, 0.f, 0.f, 0.f};
                v8f vacc = {0.f, 0.f, 0.f, 0.f, 0.f, 0.f, 0.f, 0.f};
#pragma unroll
                for (int c = 0; c < 2; ++c) {
                    // transient template A-fragment (8 VGPRs)
                    v16bf at, bk, bv;
#pragma unroll
                    for (int j = 0; j < 8; ++j) {
                        const int K0 = 32 * c + 16 * g + 2 * j;
                        v2f p = *(const v2f*)(s_t + (t * 16 + nm) * 64 + K0);
                        at[2 * j]     = (__bf16)p.x;
                        at[2 * j + 1] = (__bf16)p.y;
                        const int k = 32 * c + 16 * g + 2 * j;
                        v2bf pk = *(const v2bf*)(s_kw + (h * 16 + nm) * 64 + k);
                        v2bf pv = *(const v2bf*)(s_vw + (h * 16 + nm) * 64 + k);
                        bk[2 * j] = pk.x; bk[2 * j + 1] = pk.y;
                        bv[2 * j] = pv.x; bv[2 * j + 1] = pv.y;
                    }
                    kacc = wmma_bf16(at, bk, kacc);
                    vacc = wmma_bf16(at, bv, vacc);
                }
                // logits: dot over d (the 16 N-lanes) via shfl butterflies,
                // then streaming softmax update (all elementwise per row r)
#pragma unroll
                for (int r = 0; r < 8; ++r) {
                    float part = qacc[r] * kacc[r];
                    part += __shfl_xor(part, 1, 16);
                    part += __shfl_xor(part, 2, 16);
                    part += __shfl_xor(part, 4, 16);
                    part += __shfl_xor(part, 8, 16);
                    const float l    = part + bias_t[t];
                    const float newm = fmaxf(mx[r], l);
                    const float corr = __expf(mx[r] - newm);
                    const float e    = __expf(l - newm);
                    sm[r]   = sm[r] * corr + e;
                    wacc[r] = wacc[r] * corr + e * vacc[r];
                    mx[r]   = newm;
                }
            }

            // normalize and park wv tile in LDS as bf16 (row = r+8g, col = h*16+d)
#pragma unroll
            for (int r = 0; r < 8; ++r)
                s_wv[(r + 8 * g) * 64 + h * 16 + nm] = (__bf16)(wacc[r] / sm[r]);
        }

        // ---- pipeline: done reading s_q/s_t -> stage next tile async -------
        if (it + 1 < TPW) {
            wait_ds0();  // ensure all LDS reads of this tile have retired
            stage_tile(qe, tpl, (tile_base + (it + 1) * WAVES + wave) * 16,
                       qdst, tdst, lane);
        }

        // ---- output projection: [16x64]x[64x128], 8 chunks x 2 WMMAs -------
        v16bf awv[2];
#pragma unroll
        for (int c = 0; c < 2; ++c) {
#pragma unroll
            for (int j = 0; j < 8; ++j) {
                const int K0 = 32 * c + 16 * g + 2 * j;
                v2bf pr = *(const v2bf*)(s_wv + nm * 64 + K0);
                awv[c][2 * j]     = pr.x;
                awv[c][2 * j + 1] = pr.y;
            }
        }
#pragma unroll 1
        for (int nc = 0; nc < 8; ++nc) {
            v8f oacc = {0.f, 0.f, 0.f, 0.f, 0.f, 0.f, 0.f, 0.f};
#pragma unroll
            for (int c = 0; c < 2; ++c) {
                v16bf b;
#pragma unroll
                for (int j = 0; j < 8; ++j) {
                    const int k = 32 * c + 16 * g + 2 * j;
                    v2bf pr = *(const v2bf*)(s_ow + (nc * 16 + nm) * 64 + k);
                    b[2 * j]     = pr.x;
                    b[2 * j + 1] = pr.y;
                }
                oacc = wmma_bf16(awv[c], b, oacc);
            }
            const float bc = ob[nc * 16 + nm];
#pragma unroll
            for (int r = 0; r < 8; ++r) {
                const int row = p0 + r + 8 * g;
                out[(size_t)row * CQ + nc * 16 + nm] = (oacc[r] + bc) * gate;
            }
        }
    }
}

extern "C" void kernel_launch(void* const* d_in, const int* in_sizes, int n_in,
                              void* d_out, int out_size, void* d_ws, size_t ws_size,
                              hipStream_t stream) {
    const float* qe   = (const float*)d_in[0];  // query_embedding
    const float* tpl  = (const float*)d_in[1];  // template_pair_representation
    const float* tmsk = (const float*)d_in[2];  // template_mask
    const float* qw   = (const float*)d_in[3];
    const float* kw   = (const float*)d_in[4];
    const float* vw   = (const float*)d_in[5];
    const float* ow   = (const float*)d_in[6];
    const float* ob   = (const float*)d_in[7];
    float* out = (float*)d_out;

    const int positions_per_block = WAVES * TPW * 16;        // 512
    const int blocks = P_TOT / positions_per_block;          // 288

    hipLaunchKernelGGL(tpa_wmma_kernel, dim3(blocks), dim3(WAVES * 32),
                       SMEM_BYTES, stream,
                       qe, tpl, tmsk, qw, kw, vw, ow, ob, out);
}
